// FreqLinearAttention_37709812859355
// MI455X (gfx1250) — compile-verified, matrix operand order checked by reading, not run on hardware
//
#include <hip/hip_runtime.h>
#include <stddef.h>

// ---------------------------------------------------------------------------
// FreqLinearAttention for MI455X (gfx1250, wave32, WMMA + TDM)
//   B=4, M=4096, D=1024, H=16, d=64, R=64 (R/2 = 32 = wave width)
// Pipeline:
//   1) fft_fwd   : 1024-pt real FFT per row (Stockham radix-2 in LDS) -> fx (bf16)
//   2) gemm x3   : q/k/v = fx @ W^T + b  via v_wmma_f32_16x16x32_bf16,
//                  B panel staged to LDS by the Tensor Data Mover, double-buffered
//   3) rff x2    : phi = [cos,sin](t @ Wrf + brf)/sqrt(32), one wave per (b,h,m)
//   4) kv_sum    : KV[64x64], Ksum[64] per (b,h)  (f32 VALU + LDS tiles)
//   5) attn      : num/den normalize -> attn (bf16)
//   6) gemm      : out2 = attn @ Wp^T + bp (WMMA + TDM)
//   7) fft_inv   : Hermitian-extend + inverse FFT -> d_out (f32)
// ---------------------------------------------------------------------------

typedef __attribute__((ext_vector_type(16))) __bf16 v16bf;
typedef __attribute__((ext_vector_type(8)))  float  v8f;
typedef __attribute__((ext_vector_type(4)))  unsigned int u32x4;
typedef __attribute__((ext_vector_type(8)))  int i32x8;
typedef __attribute__((ext_vector_type(4)))  int i32x4;

union Frag16 { v16bf v; uint4 q[2]; };

__device__ inline unsigned short f32_to_bf16(float f) {
    unsigned u = __float_as_uint(f);
    u += 0x7FFFu + ((u >> 16) & 1u);      // round-to-nearest-even
    return (unsigned short)(u >> 16);
}

// ---------------- f32 -> bf16 elementwise convert (weights) ----------------
__global__ void __launch_bounds__(256)
f2bf_kernel(const float* __restrict__ src, unsigned short* __restrict__ dst, int n) {
    int i = blockIdx.x * blockDim.x + threadIdx.x;
    if (i < n) dst[i] = f32_to_bf16(src[i]);
}

// ---------------- 1024-point Stockham radix-2 complex FFT in LDS -----------
__device__ inline float2* fft1024_stages(float2* sA, float2* sB, int tid, float sign) {
    float2* src = sA;
    float2* dst = sB;
    for (int Ns = 1; Ns < 1024; Ns <<= 1) {
        __syncthreads();
        for (int j = tid; j < 512; j += 256) {
            int jm   = j & (Ns - 1);
            int base = ((j - jm) << 1) + jm;     // (j/Ns)*2Ns + jm
            float2 a = src[j];
            float2 b = src[j + 512];
            float ang = sign * 3.14159265358979323846f * (float)jm / (float)Ns;
            float s, c;
            __sincosf(ang, &s, &c);
            float tx = b.x * c - b.y * s;
            float ty = b.x * s + b.y * c;
            dst[base].x      = a.x + tx;  dst[base].y      = a.y + ty;
            dst[base + Ns].x = a.x - tx;  dst[base + Ns].y = a.y - ty;
        }
        float2* t = src; src = dst; dst = t;
    }
    __syncthreads();
    return src;
}

__global__ void __launch_bounds__(256)
fft_fwd_kernel(const float* __restrict__ x, unsigned short* __restrict__ fx) {
    __shared__ float2 sA[1024];
    __shared__ float2 sB[1024];
    const int row = blockIdx.x;
    const int tid = threadIdx.x;
    const float* xr = x + (size_t)row * 1024;
    for (int i = tid; i < 1024; i += 256) { sA[i].x = xr[i]; sA[i].y = 0.0f; }
    float2* res = fft1024_stages(sA, sB, tid, -1.0f);
    unsigned short* fr = fx + (size_t)row * 1024;
    const float invN = 1.0f / 1024.0f;
    for (int i = tid; i < 1024; i += 256) {
        float val = (i <= 512) ? res[i].x * invN : (-res[i - 513].y * invN);
        fr[i] = f32_to_bf16(val);
    }
}

__global__ void __launch_bounds__(256)
fft_inv_kernel(const float* __restrict__ y, float* __restrict__ out) {
    __shared__ float2 sA[1024];
    __shared__ float2 sB[1024];
    const int row = blockIdx.x;
    const int tid = threadIdx.x;
    const float* yr = y + (size_t)row * 1024;
    for (int i = tid; i < 1024; i += 256) {
        float re, im;
        if (i < 512)       { re = yr[i];          im = -yr[512 + i]; }
        else if (i == 512) { re = 0.0f;           im = 0.0f;         }
        else { int kk = 1024 - i; re = yr[kk];    im = yr[512 + kk]; } // conj(c[kk])
        sA[i].x = re; sA[i].y = im;
    }
    float2* res = fft1024_stages(sA, sB, tid, 1.0f);
    float* orow = out + (size_t)row * 1024;
    for (int i = tid; i < 1024; i += 256) orow[i] = res[i].x;
}

// ---------------- Tensor Data Mover: 2D bf16 tile -> LDS -------------------
// Loads a (64 rows x 64 elems) bf16 tile (row stride = dim0 elems) into LDS.
// D# built per CDNA5 ISA 8.3/8.4: group0 = {count,lds,gaddr,type=2},
// group1 = {data_size=2B, tensor dims, tile 64x64, dim0 stride}.
__device__ __forceinline__ void tdm_load_tile64x64_bf16(
        unsigned ldsoff, const unsigned short* gptr, int dim0, int dim1) {
    unsigned long long ga = (unsigned long long)gptr;
    u32x4 g0 = {};
    g0[0] = 1u;                                        // count=1, user mode
    g0[1] = ldsoff;                                    // lds_addr
    g0[2] = (unsigned)ga;                              // global_addr[31:0]
    g0[3] = (unsigned)((ga >> 32) & 0x1FFFFFFu) | (2u << 30); // addr[56:32] | type=2
    i32x8 g1 = {};
    g1[0] = (int)(1u << 16);                           // data_size=1 -> 2 bytes
    g1[1] = (int)(((unsigned)dim0 & 0xFFFFu) << 16);   // tensor_dim0[15:0]
    g1[2] = (int)((((unsigned)dim0 >> 16) & 0xFFFFu)
                | (((unsigned)dim1 & 0xFFFFu) << 16)); // dim0[31:16] | dim1[15:0]
    g1[3] = (int)(((unsigned)dim1 >> 16)
                | (64u << 16));                        // dim1[31:16] | tile_dim0=64
    g1[4] = 64;                                        // tile_dim1=64, tile_dim2=0
    g1[5] = dim0;                                      // tensor_dim0_stride[31:0]
    g1[6] = 0;                                         // stride hi | dim1_stride lo
    g1[7] = 0;
    i32x4 z4 = {};
#if defined(__clang_major__) && (__clang_major__ >= 23)
    i32x8 z8 = {};
    __builtin_amdgcn_tensor_load_to_lds(g0, g1, z4, z4, z8, 0);
#else
    __builtin_amdgcn_tensor_load_to_lds(g0, g1, z4, z4, 0);
#endif
}

// LDS aperture keeps the byte offset in the low 32 bits of the generic address.
__device__ __forceinline__ unsigned lds_byte_offset(const void* p) {
    return (unsigned)(unsigned long long)p;
}

// ---------------- bf16 WMMA GEMM: C = A @ Bw^T + bias ----------------------
// A: (rows x K) bf16 row-major; Bw: (N x K) bf16 row-major.
// Block = 8 waves -> 128(M) x 64(N); K consumed in 64-wide chunks.
// B panel chunk (64 rows x 64 K elems, 8 KB) DMA'd to LDS by the TDM,
// double-buffered; wave 0 issues DMA for chunk i+1 while all compute chunk i.
#define KCH 64
__global__ void __launch_bounds__(256)
gemm_bf16_wmma(const unsigned short* __restrict__ A,
               const unsigned short* __restrict__ Bw,
               const float* __restrict__ bias,
               float* __restrict__ C,
               int rows, int N, int K) {
    (void)rows;
    __shared__ unsigned short btile[2][64 * KCH];      // 16 KB double buffer

    const int lane   = threadIdx.x & 31;
    const int wave   = threadIdx.x >> 5;
    const int ntiles = N >> 6;
    const int n0 = (blockIdx.x % ntiles) << 6;
    const int m0 = ((blockIdx.x / ntiles) << 7) + (wave << 4);

    // ISA 16-bit A layout: lane<16 -> K{0..7,16..23}; lane>=16 -> K{8..15,24..31}
    const int krow  = lane & 15;
    const int khalf = (lane >> 4) << 3;                // 0 or 8 (elements)

    const unsigned short* aptr  = A  + (size_t)(m0 + krow) * K + khalf;
    const unsigned short* bpan  = Bw + (size_t)n0 * K;  // 64-row column panel
    const unsigned lds0 = lds_byte_offset(&btile[0][0]);
    const unsigned lds1 = lds_byte_offset(&btile[1][0]);

    const int nch = K / KCH;
    if (wave == 0) {
        tdm_load_tile64x64_bf16(lds0, bpan, K, N);
        __builtin_amdgcn_s_wait_tensorcnt(0);
    }
    __syncthreads();

    v8f acc[4] = {};
    for (int ch = 0; ch < nch; ++ch) {
        if ((ch + 1 < nch) && wave == 0)               // prefetch next chunk via TDM
            tdm_load_tile64x64_bf16((ch & 1) ? lds0 : lds1,
                                    bpan + (size_t)(ch + 1) * KCH, K, N);

        const unsigned short* bt = &btile[ch & 1][0];
#pragma unroll
        for (int ks = 0; ks < 2; ++ks) {               // two 32-wide K steps
            const int kg = ch * KCH + ks * 32;
            Frag16 a;
            a.q[0] = *(const uint4*)(aptr + kg);
            a.q[1] = *(const uint4*)(aptr + kg + 16);
#pragma unroll
            for (int j = 0; j < 4; ++j) {
                Frag16 b;
                const unsigned short* p =
                    bt + (size_t)((j << 4) + krow) * KCH + ks * 32 + khalf;
                b.q[0] = *(const uint4*)(p);
                b.q[1] = *(const uint4*)(p + 16);
                acc[j] = __builtin_amdgcn_wmma_f32_16x16x32_bf16(
                    false, a.v, false, b.v, (short)0, acc[j], false, false);
            }
        }

        if (ch + 1 < nch) {
            if (wave == 0) __builtin_amdgcn_s_wait_tensorcnt(0);
            __syncthreads();                           // publish next buffer
        }
    }

    // C layout: VGPR i -> M = m0 + i + 8*(lane>=16), N = n0 + 16*j + (lane&15)
    const int rbase = m0 + ((lane >> 4) << 3);
#pragma unroll
    for (int j = 0; j < 4; ++j) {
        const int cn = n0 + (j << 4) + (lane & 15);
        const float bi = bias[cn];
#pragma unroll
        for (int i = 0; i < 8; ++i)
            C[(size_t)(rbase + i) * N + cn] = acc[j][i] + bi;
    }
}

// ---------------- RFF: one wave per (b,h,m); lane = frequency r ------------
__global__ void __launch_bounds__(256)
rff_kernel(const float* __restrict__ t,     // (B*M, 1024) rows; head slice h*64
           const float* __restrict__ Wrf,   // (H, 64, 32)
           const float* __restrict__ brf,   // (H, 32)
           float* __restrict__ phi,         // (B*H*M, 64)
           int Mn, int Hn) {
    const int lane = threadIdx.x & 31;
    const int gw = blockIdx.x * 8 + (threadIdx.x >> 5);   // (b*H + h)*M + m
    const int m  = gw % Mn;
    const int bh = gw / Mn;
    const int h  = bh % Hn;
    const int b  = bh / Hn;
    const float* trow = t + (size_t)(b * Mn + m) * 1024 + h * 64;
    const float* wcol = Wrf + (size_t)h * 64 * 32 + lane;
    float proj = brf[h * 32 + lane];
#pragma unroll 8
    for (int j = 0; j < 64; ++j)
        proj = fmaf(trow[j], wcol[(size_t)j * 32], proj);
    const float sc = 0.17677669529663687f;  // 1/sqrt(32)
    float cs = __cosf(proj) * sc;
    float sn = __sinf(proj) * sc;
    float* orow = phi + (size_t)gw * 64;
    orow[lane]      = cs;
    orow[32 + lane] = sn;
}

// ---------------- KV[r,d] = sum_m phi_k*v ; Ksum[r] = sum_m phi_k ----------
__global__ void __launch_bounds__(256)
kv_kernel(const float* __restrict__ phi_k,  // (B*H, M, 64)
          const float* __restrict__ v,      // (B*M, 1024) rows; head slice h*64
          float* __restrict__ KV,           // (B*H, 64, 64)
          float* __restrict__ Ksum,         // (B*H, 64)
          int Mn, int Hn) {
    const int bh = blockIdx.x;
    const int b = bh / Hn, h = bh % Hn;
    const int t = threadIdx.x;
    const int r  = t >> 2;             // 0..63
    const int dg = (t & 3) << 4;       // 0,16,32,48
    __shared__ float sp[16][64];
    __shared__ float sv[16][64];
    float acc[16] = {};
    float ks = 0.0f;
    const float* prow = phi_k + (size_t)bh * Mn * 64;
    for (int m0 = 0; m0 < Mn; m0 += 16) {
        __syncthreads();
        for (int idx = t; idx < 16 * 64; idx += 256) {
            int mm = idx >> 6, cc = idx & 63;
            sp[mm][cc] = prow[(size_t)(m0 + mm) * 64 + cc];
            sv[mm][cc] = v[(size_t)(b * Mn + m0 + mm) * 1024 + h * 64 + cc];
        }
        __syncthreads();
#pragma unroll
        for (int mm = 0; mm < 16; ++mm) {
            float p = sp[mm][r];
            if ((t & 3) == 0) ks += p;
#pragma unroll
            for (int i = 0; i < 16; ++i) acc[i] = fmaf(p, sv[mm][dg + i], acc[i]);
        }
    }
    float* kvout = KV + (size_t)bh * 4096 + r * 64 + dg;
#pragma unroll
    for (int i = 0; i < 16; ++i) kvout[i] = acc[i];
    if ((t & 3) == 0) Ksum[bh * 64 + r] = ks;
}

// ---------------- num/den normalize -> attn (bf16) -------------------------
__global__ void __launch_bounds__(256)
attn_kernel(const float* __restrict__ phi_q, const float* __restrict__ KV,
            const float* __restrict__ Ksum, unsigned short* __restrict__ attn_bf,
            int Mn, int Hn) {
    const int lane = threadIdx.x & 31;
    const int gw = blockIdx.x * 8 + (threadIdx.x >> 5);   // (b*H + h)*M + m
    const int m  = gw % Mn;
    const int bh = gw / Mn;
    const int b = bh / Hn, h = bh % Hn;
    const float* pq   = phi_q + (size_t)gw * 64;
    const float* kv   = KV + (size_t)bh * 4096;
    const float* ksum = Ksum + bh * 64;
    float n0 = 0.0f, n1 = 0.0f, den = 0.0f;
#pragma unroll 4
    for (int r = 0; r < 64; ++r) {
        float p = pq[r];
        n0  = fmaf(p, kv[r * 64 + lane],      n0);
        n1  = fmaf(p, kv[r * 64 + 32 + lane], n1);
        den = fmaf(p, ksum[r],                den);
    }
    float inv = 1.0f / (den + 1e-6f);
    unsigned short* orow = attn_bf + (size_t)(b * Mn + m) * 1024 + h * 64;
    orow[lane]      = f32_to_bf16(n0 * inv);
    orow[32 + lane] = f32_to_bf16(n1 * inv);
}

// ---------------------------------------------------------------------------
extern "C" void kernel_launch(void* const* d_in, const int* in_sizes, int n_in,
                              void* d_out, int out_size, void* d_ws, size_t ws_size,
                              hipStream_t stream) {
    (void)in_sizes; (void)n_in; (void)out_size; (void)ws_size;
    const float* x   = (const float*)d_in[0];
    const float* Wq  = (const float*)d_in[1];
    const float* bq  = (const float*)d_in[2];
    const float* Wk  = (const float*)d_in[3];
    const float* bk  = (const float*)d_in[4];
    const float* Wv  = (const float*)d_in[5];
    const float* bv  = (const float*)d_in[6];
    const float* Wp  = (const float*)d_in[7];
    const float* bp  = (const float*)d_in[8];
    const float* Wrf = (const float*)d_in[9];
    const float* brf = (const float*)d_in[10];
    float* out = (float*)d_out;

    char* ws = (char*)d_ws;
    const size_t MB = 1ull << 20;
    unsigned short* fx_bf   = (unsigned short*)(ws + 0);        // 32 MB
    unsigned short* Wq_bf   = (unsigned short*)(ws + 32  * MB); // 2 MB each
    unsigned short* Wk_bf   = (unsigned short*)(ws + 34  * MB);
    unsigned short* Wv_bf   = (unsigned short*)(ws + 36  * MB);
    unsigned short* Wp_bf   = (unsigned short*)(ws + 38  * MB);
    float*          q       = (float*)        (ws + 40  * MB); // 64 MB each
    float*          kbuf    = (float*)        (ws + 104 * MB);
    float*          vbuf    = (float*)        (ws + 168 * MB);
    float*          phi_q   = (float*)        (ws + 232 * MB);
    float*          phi_k   = (float*)        (ws + 296 * MB);
    float*          KV      = (float*)        (ws + 360 * MB); // 1 MB
    float*          Ksum    = (float*)        (ws + 361 * MB); // 16 KB
    unsigned short* attn_bf = (unsigned short*)(ws + 362 * MB); // 32 MB
    float*          out2    = (float*)        (ws + 394 * MB); // 64 MB

    const int ROWS = 4 * 4096;       // B*M
    const int Mn = 4096, Hn = 16;
    const int WD = 1024 * 1024;

    f2bf_kernel<<<(WD + 255) / 256, 256, 0, stream>>>(Wq, Wq_bf, WD);
    f2bf_kernel<<<(WD + 255) / 256, 256, 0, stream>>>(Wk, Wk_bf, WD);
    f2bf_kernel<<<(WD + 255) / 256, 256, 0, stream>>>(Wv, Wv_bf, WD);
    f2bf_kernel<<<(WD + 255) / 256, 256, 0, stream>>>(Wp, Wp_bf, WD);

    fft_fwd_kernel<<<ROWS, 256, 0, stream>>>(x, fx_bf);

    dim3 ggrid((ROWS / 128) * (1024 / 64));
    gemm_bf16_wmma<<<ggrid, 256, 0, stream>>>(fx_bf, Wq_bf, bq, q,    ROWS, 1024, 1024);
    gemm_bf16_wmma<<<ggrid, 256, 0, stream>>>(fx_bf, Wk_bf, bk, kbuf, ROWS, 1024, 1024);
    gemm_bf16_wmma<<<ggrid, 256, 0, stream>>>(fx_bf, Wv_bf, bv, vbuf, ROWS, 1024, 1024);

    const int NW = 4 * 16 * 4096;    // B*H*M waves
    rff_kernel<<<NW / 8, 256, 0, stream>>>(q,    Wrf, brf, phi_q, Mn, Hn);
    rff_kernel<<<NW / 8, 256, 0, stream>>>(kbuf, Wrf, brf, phi_k, Mn, Hn);

    kv_kernel<<<4 * 16, 256, 0, stream>>>(phi_k, vbuf, KV, Ksum, Mn, Hn);
    attn_kernel<<<NW / 8, 256, 0, stream>>>(phi_q, KV, Ksum, attn_bf, Mn, Hn);

    gemm_bf16_wmma<<<ggrid, 256, 0, stream>>>(attn_bf, Wp_bf, bp, out2, ROWS, 1024, 1024);

    fft_inv_kernel<<<ROWS, 256, 0, stream>>>(out2, out);
}
